// _Block_20143396618412
// MI455X (gfx1250) — matrix operand order repaired and checked by reading, not compiled, not run on hardware
//
#include <hip/hip_runtime.h>
#include <hip/hip_bf16.h>

// ---------------------------------------------------------------------------
// ConvNeXt block for MI455X (gfx1250): dwconv7x7 + LN + MLP(128->512->128,
// exact GELU) + layerscale + residual.  GEMMs use v_wmma_f32_16x16x32_bf16.
// ---------------------------------------------------------------------------

#define DIMC   128
#define HID    512
#define HWDIM  56
#define NBATCH 64
#define MTOT   (NBATCH * HWDIM * HWDIM)   // 200704 pixels, divisible by 128

typedef __attribute__((ext_vector_type(16))) __bf16 v16bf;
typedef __attribute__((ext_vector_type(8)))  float  v8f;

// ---------------------------------------------------------------------------
// WMMA 16x16x32 bf16 fragment layouts (ISA 7.12.2, wave32):
//  A (16xK=32): lane<16 holds row m=lane, k locals {0..7,16..23} in j=0..15;
//               lane>=16 holds row m=lane-16, k locals {8..15,24..31}.
//  B (K=32x16): lane<16 holds col n=lane, k locals {0..7,16..23} in j=0..15;
//               lane>=16 holds col n=lane-16, k locals {8..15,24..31} (same as A).
//  C/D f32:     lane holds col n=lane%16; vgpr v holds row m=v+8*(lane>=16).
// We store A/B pre-swizzled so each lane loads 16 contiguous bf16 (32 B).
// ---------------------------------------------------------------------------

// branchless inverse A mapping: k_local -> (hi, j)
//  kl 0..7  -> hi 0, j = kl      ; kl 8..15  -> hi 1, j = kl-8
//  kl 16..23-> hi 0, j = kl-8    ; kl 24..31 -> hi 1, j = kl-16
__device__ __forceinline__ int a_hi(int kl) { return (kl >> 3) & 1; }
__device__ __forceinline__ int a_j (int kl) { return (kl & 7) | ((kl >> 1) & 8); }

__device__ __forceinline__ int a_elem_index(int mt, int nkt, int kt, int r, int kl) {
    return (((mt * nkt + kt) * 32) + r + 16 * a_hi(kl)) * 16 + a_j(kl);
}

// inverse B mapping: element (col n, k_local kl) -> flat index inside tiles
__device__ __forceinline__ int b_elem_index(int nt, int nkt, int kt, int n, int kl) {
    return (((nt * nkt + kt) * 32) + n + 16 * a_hi(kl)) * 16 + a_j(kl);
}

// branchless exact-GELU: Abramowitz-Stegun 7.1.26 erf, |err| < 1.5e-7.
// Raw v_rcp_f32 (1 ULP) + v_exp_f32 + Horner chain; no EXEC divergence and
// no correctly-rounded-divide expansion (those TRANS ops co-execute with the
// XDL/WMMA pipe).
__device__ __forceinline__ float gelu_exact(float v) {
    const float xa = fabsf(v) * 0.70710678118f;          // |x|/sqrt(2)
    const float t  = __builtin_amdgcn_rcpf(1.0f + 0.3275911f * xa);
    const float p  = t * (0.254829592f +
                     t * (-0.284496736f +
                     t * (1.421413741f +
                     t * (-1.453152027f +
                     t * 1.061405429f))));
    const float er = copysignf(1.0f - p * __expf(-xa * xa), v);
    return 0.5f * v * (1.0f + er);
}

// ---------------------------------------------------------------------------
// Kernel 0: pack w1 [512x128] and w2 [128x512] into bf16 B-fragment layout.
// ---------------------------------------------------------------------------
__global__ void pack_weights(const float* __restrict__ w1,
                             const float* __restrict__ w2,
                             __bf16* __restrict__ b1s,
                             __bf16* __restrict__ b2s) {
    int idx = blockIdx.x * 256 + threadIdx.x;
    if (idx < HID * DIMC) {
        // GEMM1 B: k = c (K=128, 4 k-tiles), n = f (512, 32 n-tiles)
        int f = idx / DIMC, c = idx % DIMC;
        b1s[b_elem_index(f >> 4, 4, c >> 5, f & 15, c & 31)] = (__bf16)w1[idx];
        // GEMM2 B: k = f (K=512, 16 k-tiles), n = c (128, 8 n-tiles)
        int c2 = idx / HID, f2 = idx % HID;
        b2s[b_elem_index(c2 >> 4, 16, f2 >> 5, c2 & 15, f2 & 31)] = (__bf16)w2[idx];
    }
}

// ---------------------------------------------------------------------------
// Kernel 1: depthwise 7x7 conv + bias + LayerNorm over C, emit bf16 A-matrix
// in swizzled fragment layout.  One block per (n,h); 8 waves x 7 w-positions;
// each lane owns 4 channels, LN reduction via wave32 xor-shuffle.
// x stays resident in the 192MB L2, so the 7x re-read across h is cheap.
// ---------------------------------------------------------------------------
__global__ void dwconv_ln(const float* __restrict__ x,
                          const float* __restrict__ dww,
                          const float* __restrict__ dwb,
                          const float* __restrict__ lnw,
                          const float* __restrict__ lnb,
                          __bf16* __restrict__ a_swz) {
    const int n    = blockIdx.x / HWDIM;
    const int h    = blockIdx.x % HWDIM;
    const int wv   = threadIdx.x >> 5;
    const int lane = threadIdx.x & 31;
    const float* xn = x + (size_t)n * DIMC * HWDIM * HWDIM;

    for (int i = 0; i < 7; ++i) {
        const int w = wv + 8 * i;           // 8 waves x 7 = 56 columns
        float s[4];
#pragma unroll
        for (int q = 0; q < 4; ++q) {
            const int c = lane * 4 + q;
            const float* xc = xn + (size_t)c * HWDIM * HWDIM;
            const float* wc = dww + c * 49;
            float acc = dwb[c];
            for (int kh = 0; kh < 7; ++kh) {
                const int hh = h + kh - 3;
                if (hh < 0 || hh >= HWDIM) continue;
#pragma unroll
                for (int kw = 0; kw < 7; ++kw) {
                    const int ww = w + kw - 3;
                    if (ww < 0 || ww >= HWDIM) continue;
                    acc += xc[hh * HWDIM + ww] * wc[kh * 7 + kw];
                }
            }
            s[q] = acc;
        }
        // LayerNorm over 128 channels (4/lane x 32 lanes)
        float sum = s[0] + s[1] + s[2] + s[3];
        float sq  = s[0]*s[0] + s[1]*s[1] + s[2]*s[2] + s[3]*s[3];
#pragma unroll
        for (int m = 16; m >= 1; m >>= 1) {
            sum += __shfl_xor(sum, m, 32);
            sq  += __shfl_xor(sq,  m, 32);
        }
        const float mu  = sum * (1.0f / 128.0f);
        const float var = sq * (1.0f / 128.0f) - mu * mu;
        const float inv = rsqrtf(var + 1e-6f);

        const int mrow = n * (HWDIM * HWDIM) + h * HWDIM + w;
        const int mt = mrow >> 4, r = mrow & 15;
#pragma unroll
        for (int q = 0; q < 4; ++q) {
            const int c = lane * 4 + q;
            const float yv = (s[q] - mu) * inv * lnw[c] + lnb[c];
            a_swz[a_elem_index(mt, 4, c >> 5, r, c & 31)] = (__bf16)yv;
        }
    }
}

// ---------------------------------------------------------------------------
// Kernel 2: fused MLP. Block = 256 thr = 8 waves; M-tile = 128 rows.
// Wave w owns rows [w*16, w*16+16): GEMM1 -> GELU -> LDS h-slice (A-swizzle,
// wave-local so no barrier) -> GEMM2 -> barrier -> LDS transpose -> coalesced
// residual store in NCHW.
// ---------------------------------------------------------------------------
__global__ void __launch_bounds__(256) mlp_block(
        const __bf16* __restrict__ a_swz,
        const __bf16* __restrict__ b1s,
        const __bf16* __restrict__ b2s,
        const float*  __restrict__ b1,
        const float*  __restrict__ b2,
        const float*  __restrict__ gamma,
        const float*  __restrict__ x,
        float*        __restrict__ out) {
    __shared__ __align__(16) char smem[131072];   // 128KB: h slices, then stage
    __bf16* hlds  = (__bf16*)smem;
    float*  stage = (float*)smem;                  // reused after barrier

    const int wv   = threadIdx.x >> 5;
    const int lane = threadIdx.x & 31;
    const int mt   = blockIdx.x * 8 + wv;          // global 16-row tile id
    const int nc   = lane & 15;                    // C/D column n
    const int vhi  = (lane >> 4) * 8;              // C/D row offset

    // ---- Phase 1: GEMM1 (M16 x K128 x N512) + bias + exact GELU ----
    v16bf a[4];
#pragma unroll
    for (int kt = 0; kt < 4; ++kt)
        a[kt] = *(const v16bf*)(a_swz + ((size_t)(mt * 4 + kt) * 32 + lane) * 16);

    __bf16* hslice = hlds + wv * 16 * HID;         // 16KB per wave

    for (int nt = 0; nt < 32; ++nt) {
        v8f acc = {};
#pragma unroll
        for (int kt = 0; kt < 4; ++kt) {
            v16bf b = *(const v16bf*)(b1s + ((size_t)(nt * 4 + kt) * 32 + lane) * 16);
            acc = __builtin_amdgcn_wmma_f32_16x16x32_bf16(
                      false, a[kt], false, b, (short)0, acc, false, false);
        }
        const int f    = nt * 16 + nc;             // hidden-unit index
        const float bf = b1[f];
        const int kt2  = f >> 5;
        const int kl   = f & 31;
        const int base = (kt2 * 32 + 16 * a_hi(kl)) * 16 + a_j(kl);  // branchless
#pragma unroll
        for (int v = 0; v < 8; ++v) {
            const float hv = gelu_exact(acc[v] + bf);
            const int rr = v + vhi;                // row within subtile
            hslice[base + rr * 16] = (__bf16)hv;
        }
    }

    // ---- Phase 2: GEMM2 (M16 x K512 x N128), wave-local h in LDS ----
    v8f acc2[8];
#pragma unroll
    for (int nt2 = 0; nt2 < 8; ++nt2) {
        v8f acc = {};
#pragma unroll 4
        for (int kt2 = 0; kt2 < 16; ++kt2) {
            v16bf a2 = *(const v16bf*)(hslice + (kt2 * 32 + lane) * 16);
            v16bf b  = *(const v16bf*)(b2s + ((size_t)(nt2 * 16 + kt2) * 32 + lane) * 16);
            acc = __builtin_amdgcn_wmma_f32_16x16x32_bf16(
                      false, a2, false, b, (short)0, acc, false, false);
        }
        acc2[nt2] = acc;
    }

    __syncthreads();   // all waves done reading h LDS before stage overwrite

    // ---- stage [c][m] transpose in LDS (pad 133 -> conflict-free) ----
    const int STM = 133;
#pragma unroll
    for (int nt2 = 0; nt2 < 8; ++nt2) {
        const int c = nt2 * 16 + nc;
#pragma unroll
        for (int v = 0; v < 8; ++v) {
            const int ml = wv * 16 + v + vhi;
            stage[c * STM + ml] = acc2[nt2][v];
        }
    }
    __syncthreads();

    // ---- coalesced epilogue: out = x + gamma*(y + b2), NCHW ----
    const int m_base = blockIdx.x * 128;
    for (int it = 0; it < 64; ++it) {
        const int e  = it * 256 + threadIdx.x;
        const int ml = e & 127;                     // consecutive across lanes
        const int c  = e >> 7;
        const int m  = m_base + ml;
        const int n  = m / (HWDIM * HWDIM);
        const int hw = m % (HWDIM * HWDIM);
        const size_t gi = ((size_t)n * DIMC + c) * (HWDIM * HWDIM) + hw;
        out[gi] = x[gi] + gamma[c] * (stage[c * STM + ml] + b2[c]);
    }
}

// ---------------------------------------------------------------------------
extern "C" void kernel_launch(void* const* d_in, const int* in_sizes, int n_in,
                              void* d_out, int out_size, void* d_ws, size_t ws_size,
                              hipStream_t stream) {
    const float* x    = (const float*)d_in[0];
    const float* dww  = (const float*)d_in[1];
    const float* dwb  = (const float*)d_in[2];
    const float* lnw  = (const float*)d_in[3];
    const float* lnb  = (const float*)d_in[4];
    const float* w1   = (const float*)d_in[5];
    const float* b1   = (const float*)d_in[6];
    const float* w2   = (const float*)d_in[7];
    const float* b2   = (const float*)d_in[8];
    const float* gam  = (const float*)d_in[9];
    float* out        = (float*)d_out;

    char* ws = (char*)d_ws;
    const size_t a_bytes = (size_t)MTOT * DIMC * sizeof(__bf16);   // ~51.4 MB
    __bf16* a_swz = (__bf16*)ws;
    __bf16* b1s   = (__bf16*)(ws + a_bytes);
    __bf16* b2s   = (__bf16*)(ws + a_bytes + (size_t)HID * DIMC * 2);

    pack_weights<<<(HID * DIMC + 255) / 256, 256, 0, stream>>>(w1, w2, b1s, b2s);
    dwconv_ln<<<NBATCH * HWDIM, 256, 0, stream>>>(x, dww, dwb, lnw, lnb, a_swz);
    mlp_block<<<MTOT / 128, 256, 0, stream>>>(a_swz, b1s, b2s, b1, b2, gam, x, out);
}
